// BiLSTM_CRF_16484084482612
// MI455X (gfx1250) — matrix operand order repaired and checked by reading, not compiled, not run on hardware
//
#include <hip/hip_runtime.h>
#include <hip/hip_bf16.h>

// Problem constants (match reference)
#define S_LEN   4096
#define E_DIM   1024
#define HH      512
#define G4      2048      // 4*HH gate rows
#define T_TAGS  5
#define START_TAG 3
#define STOP_TAG  4
#define NEG_INF  -10000.0f
#define NB_DIR   16       // scan blocks per direction

typedef __attribute__((ext_vector_type(16))) _Float16 v16h;
typedef __attribute__((ext_vector_type(8)))  _Float16 v8h;
typedef __attribute__((ext_vector_type(8)))  float    v8f;

union AF { v16h v; v8h h[2]; };
union HU { unsigned u; _Float16 h[2]; };

// gfx1250 async global->LDS path (guarded: falls back to plain load/store)
#if __has_builtin(__builtin_amdgcn_global_load_async_to_lds_b32) && \
    __has_builtin(__builtin_amdgcn_s_wait_asynccnt)
#define HAVE_ASYNC_LDS 1
#endif
typedef __attribute__((address_space(1))) int gas_int;   // global (AS1)
typedef __attribute__((address_space(3))) int las_int;   // LDS (AS3)

// ---------------------------------------------------------------------------
// Init: zero the grid-barrier counters (workspace is poisoned 0xAA).
__global__ void init_cnt_kernel(unsigned* __restrict__ cnt) {
    if (threadIdx.x < 2) cnt[threadIdx.x] = 0u;
}

// ---------------------------------------------------------------------------
// Embedding gather + f32 -> f16 convert: x16[s][e] = (f16) embed[sentence[s]][e]
__global__ __launch_bounds__(256)
void gather_embed_kernel(const float* __restrict__ embed,
                         const int* __restrict__ sentence,
                         _Float16* __restrict__ x16) {
    const int s = blockIdx.x;
    const int row = sentence[s];
    const float* src = embed + (size_t)row * E_DIM;
    _Float16* dst = x16 + (size_t)s * E_DIM;
    for (int e = threadIdx.x; e < E_DIM; e += 256)
        dst[e] = (_Float16)src[e];
}

// Generic f32 -> f16 convert
__global__ __launch_bounds__(256)
void cvt_f16_kernel(const float* __restrict__ src, _Float16* __restrict__ dst, int n) {
    int i = blockIdx.x * 256 + threadIdx.x;
    if (i < n) dst[i] = (_Float16)src[i];
}

// ---------------------------------------------------------------------------
// Input projection: xp[dir][s][g] = x[s] . w_ih[dir][g] + b[dir][g]
// WMMA f32 = f16 x f16.  Block = 256 thr = 8 waves; each wave owns a 16x32
// output strip (two 16x16 N tiles sharing one A fragment -> halves A traffic);
// grid = (S/16, (G4/16)/(8*2), 2 dirs).  K loop = 1024 in steps of 32.
__global__ __launch_bounds__(256)
void input_proj_wmma_kernel(const _Float16* __restrict__ x16,
                            const _Float16* __restrict__ w16base,
                            const float* __restrict__ b_f,
                            const float* __restrict__ b_b,
                            float* __restrict__ xpbase) {
    const int dir = blockIdx.z;
    const _Float16* w16 = w16base + (size_t)dir * G4 * E_DIM;
    const float* bias = dir ? b_b : b_f;
    float* xp = xpbase + (size_t)dir * S_LEN * G4;

    const int wave = threadIdx.x >> 5;
    const int lane = threadIdx.x & 31;
    const int l = lane & 15;       // row-within-tile selector
    const int g = lane >> 4;       // lane-group (K half select)

    const int m0 = blockIdx.x * 16;
    const int n0 = (blockIdx.y * 16 + wave * 2) * 16;   // two adjacent N tiles

    const _Float16* xrow  = x16 + (size_t)(m0 + l) * E_DIM;
    const _Float16* wrow0 = w16 + (size_t)(n0 + l) * E_DIM;
    const _Float16* wrow1 = wrow0 + (size_t)16 * E_DIM;

    v8f acc0 = {};
    v8f acc1 = {};
#pragma unroll 4
    for (int k0 = 0; k0 < E_DIM; k0 += 32) {
        __builtin_prefetch(xrow + k0 + 256, 0, 1);
        __builtin_prefetch(wrow0 + k0 + 256, 0, 1);
        __builtin_prefetch(wrow1 + k0 + 256, 0, 1);
        // A-frag (16x32 f16): lane<16 -> K {0..15}; lane>=16 -> K {16..31}
        AF a;
        a.h[0] = *(const v8h*)(xrow + k0 + g * 16);
        a.h[1] = *(const v8h*)(xrow + k0 + g * 16 + 8);
        // B-frags (32x16 f16): half i <-> K = 16*g + i, N = lane%16 (B[k][n] = w[n][k])
        AF b0, b1;
        b0.h[0] = *(const v8h*)(wrow0 + k0 + g * 16);
        b0.h[1] = *(const v8h*)(wrow0 + k0 + g * 16 + 8);
        b1.h[0] = *(const v8h*)(wrow1 + k0 + g * 16);
        b1.h[1] = *(const v8h*)(wrow1 + k0 + g * 16 + 8);
        acc0 = __builtin_amdgcn_wmma_f32_16x16x32_f16(false, a.v, false, b0.v,
                                                      (short)0, acc0, false, false);
        acc1 = __builtin_amdgcn_wmma_f32_16x16x32_f16(false, a.v, false, b1.v,
                                                      (short)0, acc1, false, false);
    }
    // D layout: col = n + lane%16, rows = m0 + 8*g + r
    const float bc0 = bias[n0 + l];
    const float bc1 = bias[n0 + 16 + l];
    const int rbase = m0 + g * 8;
#pragma unroll
    for (int r = 0; r < 8; ++r) {
        xp[(size_t)(rbase + r) * G4 + (n0 + l)]      = acc0[r] + bc0;
        xp[(size_t)(rbase + r) * G4 + (n0 + 16 + l)] = acc1[r] + bc1;
    }
}

// ---------------------------------------------------------------------------
// Device-scope lockstep barrier helper (monotonic counter, per direction).
__device__ inline void grid_barrier(unsigned* cnt, unsigned target) {
    __syncthreads();
    if (threadIdx.x == 0) {
        __threadfence();
        atomicAdd(cnt, 1u);
        while (__hip_atomic_load(cnt, __ATOMIC_ACQUIRE, __HIP_MEMORY_SCOPE_AGENT) < target)
            __builtin_amdgcn_s_sleep(1);
    }
    __syncthreads();
}

// ---------------------------------------------------------------------------
// Sequential LSTM recurrence, both directions concurrently.
// 32 blocks x 512 threads.  Block (dir, blk) owns Hh-slice j0 = blk*32 (32 j's
// -> 128 gate rows of w_hh).  Each thread holds its 128-half w_hh slice in
// VGPRs (64 dwords) for the whole 4096-step scan; h is broadcast in f32 via L2
// (2 KB/step) with a 16-WGP lockstep barrier per step, staged into LDS with
// the gfx1250 async global->LDS path when available.
__global__ __launch_bounds__(512)
void lstm_scan_kernel(const _Float16* __restrict__ whh16base, // [2][G4][HH] f16
                      const float* __restrict__ xpbase,       // [2][S][G4]
                      const float* __restrict__ h0,           // [2][HH]
                      const float* __restrict__ c0,           // [2][HH]
                      float* __restrict__ hbufF,              // [2 parity][2 dir][HH] f32
                      float* __restrict__ hs,                 // [2 dir][S][HH]
                      unsigned* __restrict__ cnt) {           // [2]
    const int dir = blockIdx.x >> 4;
    const int blk = blockIdx.x & 15;
    const int j0 = blk * 32;
    const int tid = threadIdx.x;
    const int r = tid >> 2;       // local gate row 0..127
    const int part = tid & 3;     // K quarter
    const int gate = r >> 5;      // 0..3 (i,f,g,o)
    const int jj = r & 31;
    const int grow = gate * HH + (j0 + jj);  // global w_hh row

    __shared__ __align__(16) float hF[HH];   // full h (f32)
    __shared__ float gLDS[128];              // gate pre-activations

    // Register-resident weight slice: 128 halves = 64 dwords (16x uint4 loads)
    const unsigned* wp = (const unsigned*)(whh16base
            + ((size_t)dir * G4 + grow) * HH + part * 128);
    unsigned wreg[64];
#pragma unroll
    for (int i = 0; i < 16; ++i) {
        uint4 q = ((const uint4*)wp)[i];
        wreg[4 * i + 0] = q.x; wreg[4 * i + 1] = q.y;
        wreg[4 * i + 2] = q.z; wreg[4 * i + 3] = q.w;
    }

    const float* xpd = xpbase + (size_t)dir * S_LEN * G4;

    // Publish initial h0 into parity-0 buffer (all blocks of a dir write the
    // same values -> benign).
    hbufF[(0 * 2 + dir) * HH + tid] = h0[dir * HH + tid];
    float c = 0.f;
    if (tid < 32) c = c0[dir * HH + j0 + tid];

    __threadfence();
    grid_barrier(&cnt[dir], NB_DIR);   // phase 0

    int parity = 0;
    for (int it = 0; it < S_LEN; ++it) {
        const int t = (dir == 0) ? it : (S_LEN - 1 - it);

        // Stage full h into LDS (async global->LDS on gfx1250)
        {
            const float* gsrcp = hbufF + (size_t)(parity * 2 + dir) * HH + tid;
#if defined(HAVE_ASYNC_LDS)
            gas_int* gsrc = (gas_int*)(unsigned long long)gsrcp;
            las_int* ldst = (las_int*)(unsigned)(unsigned long long)&hF[tid];
            __builtin_amdgcn_global_load_async_to_lds_b32(gsrc, ldst, 0, 0);
            __builtin_amdgcn_s_wait_asynccnt(0);
#else
            hF[tid] = *gsrcp;
#endif
        }
        __syncthreads();

        // Partial dot: 128 MACs per thread from register-resident weights
        const float2* hv2 = (const float2*)hF;
        float sum = 0.f;
#pragma unroll 16
        for (int i = 0; i < 64; ++i) {
            HU w; w.u = wreg[i];
            float2 hv = hv2[part * 64 + i];
            sum = fmaf((float)w.h[0], hv.x, sum);
            sum = fmaf((float)w.h[1], hv.y, sum);
        }
        // Reduce the 4 K-quarters (adjacent lanes in the same wave)
        sum += __shfl_xor(sum, 1, 32);
        sum += __shfl_xor(sum, 2, 32);
        if (part == 0) gLDS[r] = sum + xpd[(size_t)t * G4 + grow];
        __syncthreads();

        // Gate nonlinearities + state update for this block's 32 j's
        if (tid < 32) {
            float gi = gLDS[tid];
            float gf = gLDS[32 + tid];
            float gc = gLDS[64 + tid];
            float go = gLDS[96 + tid];
            float ig = 1.f / (1.f + __expf(-gi));
            float fg = 1.f / (1.f + __expf(-gf));
            float gg = tanhf(gc);
            float og = 1.f / (1.f + __expf(-go));
            c = fg * c + ig * gg;
            float h = og * tanhf(c);
            hs[((size_t)dir * S_LEN + t) * HH + j0 + tid] = h;
            hbufF[((size_t)(parity ^ 1) * 2 + dir) * HH + j0 + tid] = h;
        }
        __threadfence();
        grid_barrier(&cnt[dir], (unsigned)NB_DIR * (unsigned)(it + 2));
        parity ^= 1;
    }
}

// ---------------------------------------------------------------------------
// feats[s][tag] = concat(hf[s], hb[s]) . W_out[tag] + b_out[tag]
// 160 threads = 5 waves; wave = tag, lanes reduce over K.
__global__ __launch_bounds__(160)
void out_proj_kernel(const float* __restrict__ hs,
                     const float* __restrict__ Wout,
                     const float* __restrict__ bout,
                     float* __restrict__ feats) {
    const int s = blockIdx.x;
    const int tag = threadIdx.x >> 5;
    const int lane = threadIdx.x & 31;
    const float* hf = hs + (size_t)s * HH;
    const float* hb = hs + (size_t)S_LEN * HH + (size_t)s * HH;
    const float* wr = Wout + (size_t)tag * (2 * HH);
    float sum = 0.f;
    for (int k = lane; k < HH; k += 32) sum = fmaf(hf[k], wr[k], sum);
    for (int k = lane; k < HH; k += 32) sum = fmaf(hb[k], wr[HH + k], sum);
#pragma unroll
    for (int m = 16; m; m >>= 1) sum += __shfl_xor(sum, m, 32);
    if (lane == 0) feats[s * T_TAGS + tag] = sum + bout[tag];
}

// ---------------------------------------------------------------------------
// Viterbi forward + backtrace (T=5, sequential over S).  One wave.
__global__ void viterbi_kernel(const float* __restrict__ feats,
                               const float* __restrict__ trans,
                               int* __restrict__ bptr,
                               float* __restrict__ out) {
    __shared__ float tr[T_TAGS * T_TAGS];
    __shared__ float fv[T_TAGS];
    __shared__ float fnew[T_TAGS];
    const int tid = threadIdx.x;
    if (tid < T_TAGS * T_TAGS) tr[tid] = trans[tid];
    if (tid < T_TAGS) fv[tid] = (tid == START_TAG) ? 0.f : NEG_INF;
    __syncthreads();
    for (int t = 0; t < S_LEN; ++t) {
        if (tid < T_TAGS) {
            float best = -3.4e38f; int bi = 0;
#pragma unroll
            for (int p = 0; p < T_TAGS; ++p) {
                float sc = fv[p] + tr[tid * T_TAGS + p];
                if (sc > best) { best = sc; bi = p; }   // first-max tie-break
            }
            fnew[tid] = best + feats[t * T_TAGS + tid];
            bptr[t * T_TAGS + tid] = bi;
        }
        __syncthreads();
        if (tid < T_TAGS) fv[tid] = fnew[tid];
        __syncthreads();
    }
    if (tid == 0) {
        float best = -3.4e38f; int bi = 0;
        for (int n = 0; n < T_TAGS; ++n) {
            float sc = fv[n] + tr[STOP_TAG * T_TAGS + n];
            if (sc > best) { best = sc; bi = n; }
        }
        out[0] = best;
        int b = bi;
        for (int t = S_LEN - 1; t >= 0; --t) {
            out[1 + t] = (float)b;
            b = bptr[t * T_TAGS + b];
        }
    }
}

// ---------------------------------------------------------------------------
extern "C" void kernel_launch(void* const* d_in, const int* in_sizes, int n_in,
                              void* d_out, int out_size, void* d_ws, size_t ws_size,
                              hipStream_t stream) {
    const float* embed  = (const float*)d_in[0];
    const float* w_ih_f = (const float*)d_in[1];
    const float* w_hh_f = (const float*)d_in[2];
    const float* b_f    = (const float*)d_in[3];
    const float* w_ih_b = (const float*)d_in[4];
    const float* w_hh_b = (const float*)d_in[5];
    const float* b_b    = (const float*)d_in[6];
    const float* W_out  = (const float*)d_in[7];
    const float* b_out  = (const float*)d_in[8];
    const float* trans  = (const float*)d_in[9];
    const float* h0     = (const float*)d_in[10];
    const float* c0     = (const float*)d_in[11];
    const int*   sent   = (const int*)d_in[12];
    float* out = (float*)d_out;

    // Workspace carve-up (256B aligned)
    char* base = (char*)d_ws;
    size_t off = 0;
    auto carve = [&](size_t bytes) -> void* {
        void* p = base + off;
        off = (off + bytes + 255) & ~(size_t)255;
        return p;
    };
    _Float16* x16   = (_Float16*)carve((size_t)S_LEN * E_DIM * 2);        // 8 MB
    _Float16* wih16 = (_Float16*)carve((size_t)2 * G4 * E_DIM * 2);       // 8 MB
    _Float16* whh16 = (_Float16*)carve((size_t)2 * G4 * HH * 2);          // 4 MB
    float*    xp    = (float*)carve((size_t)2 * S_LEN * G4 * 4);          // 64 MB
    float*    hs    = (float*)carve((size_t)2 * S_LEN * HH * 4);          // 16 MB
    float*    hbufF = (float*)carve((size_t)2 * 2 * HH * 4);              // 8 KB
    float*    feats = (float*)carve((size_t)S_LEN * T_TAGS * 4);
    int*      bptr  = (int*)carve((size_t)S_LEN * T_TAGS * 4);
    unsigned* cnt   = (unsigned*)carve(256);
    (void)ws_size; (void)in_sizes; (void)n_in; (void)out_size;

    // 0) reset barrier counters (fresh every call -> graph-replay safe)
    hipLaunchKernelGGL(init_cnt_kernel, dim3(1), dim3(32), 0, stream, cnt);

    // 1) embedding gather -> f16
    hipLaunchKernelGGL(gather_embed_kernel, dim3(S_LEN), dim3(256), 0, stream,
                       embed, sent, x16);

    // 2) weight conversions -> f16
    {
        int n = G4 * E_DIM;
        hipLaunchKernelGGL(cvt_f16_kernel, dim3((n + 255) / 256), dim3(256), 0, stream,
                           w_ih_f, wih16, n);
        hipLaunchKernelGGL(cvt_f16_kernel, dim3((n + 255) / 256), dim3(256), 0, stream,
                           w_ih_b, wih16 + (size_t)G4 * E_DIM, n);
        int m = G4 * HH;
        hipLaunchKernelGGL(cvt_f16_kernel, dim3((m + 255) / 256), dim3(256), 0, stream,
                           w_hh_f, whh16, m);
        hipLaunchKernelGGL(cvt_f16_kernel, dim3((m + 255) / 256), dim3(256), 0, stream,
                           w_hh_b, whh16 + (size_t)G4 * HH, m);
    }

    // 3) big WMMA GEMM: xp[dir] = x @ w_ih[dir]^T + b[dir]
    hipLaunchKernelGGL(input_proj_wmma_kernel,
                       dim3(S_LEN / 16, (G4 / 16) / 16, 2), dim3(256), 0, stream,
                       x16, wih16, b_f, b_b, xp);

    // 4) sequential bidirectional LSTM scan (16 WGPs per direction, lockstep)
    hipLaunchKernelGGL(lstm_scan_kernel, dim3(2 * NB_DIR), dim3(512), 0, stream,
                       whh16, xp, h0, c0, hbufF, hs, cnt);

    // 5) output projection
    hipLaunchKernelGGL(out_proj_kernel, dim3(S_LEN), dim3(160), 0, stream,
                       hs, W_out, b_out, feats);

    // 6) Viterbi + backtrace -> d_out = [score, path(4096)]
    hipLaunchKernelGGL(viterbi_kernel, dim3(1), dim3(32), 0, stream,
                       feats, trans, bptr, out);
}
